// TritonDynamicAttention_71760313581858
// MI455X (gfx1250) — matrix operand order repaired and checked by reading, not compiled, not run on hardware
//
#include <hip/hip_runtime.h>

#define B_  2
#define H_  16
#define S_  2048
#define D_  64
#define BLK 32
#define NB  (S_ / BLK)   // 64

typedef __attribute__((ext_vector_type(16))) __bf16 v16bf;
typedef __attribute__((ext_vector_type(8)))  float  v8f;

union BF16Frag {
    v16bf  v;
    __bf16 b[16];
};

// ---------------------------------------------------------------------------
// Kernel 1: mask[H,S,S] -> block_mask[H,NB,NB] (int 0/1).
// Upper-triangular blocks are never used (causal) -> skip their 4KB reads.
// ---------------------------------------------------------------------------
__global__ void __launch_bounds__(256)
block_mask_kernel(const int* __restrict__ mask,
                  const float* __restrict__ bias,
                  int* __restrict__ bm) {
    const int idx = blockIdx.x;            // H*NB*NB
    const int h   = idx / (NB * NB);
    const int rem = idx % (NB * NB);
    const int ibr = rem / NB;
    const int jbr = rem % NB;

    if (jbr > ibr) {                       // causal: block never used
        if (threadIdx.x == 0) bm[idx] = 0;
        return;
    }

    const int* base = mask + (size_t)h * S_ * S_
                           + (size_t)ibr * BLK * S_
                           + (size_t)jbr * BLK;
    const int r  = threadIdx.x >> 3;        // 0..31
    const int c4 = (threadIdx.x & 7) * 4;   // 0..28
    const int4 v = *(const int4*)(base + (size_t)r * S_ + c4);
    int s = v.x + v.y + v.z + v.w;

    __shared__ int red[256];
    red[threadIdx.x] = s;
    __syncthreads();
    #pragma unroll
    for (int off = 128; off > 0; off >>= 1) {
        if ((int)threadIdx.x < off) red[threadIdx.x] += red[threadIdx.x + off];
        __syncthreads();
    }
    if (threadIdx.x == 0)
        bm[idx] = (((float)red[0] + bias[h]) > 0.0f) ? 1 : 0;
}

// ---------------------------------------------------------------------------
// Kernel 2: block-sparse causal flash attention, one wave32 per 32-row block.
// ---------------------------------------------------------------------------

// Async memory->LDS copy of 16 bytes at a constant offset (added to BOTH the
// global and LDS address per CDNA5 ISA §10.7/15.18). Tracked on ASYNCcnt.
#define ASYNC_CP16(OFF)                                                       \
    asm volatile("global_load_async_to_lds_b128 %0, %1, off offset:" #OFF     \
                 :: "v"(lds_dst), "v"(gsrc) : "memory")

__global__ void __launch_bounds__(32)
attn_kernel(const float* __restrict__ Q,
            const float* __restrict__ K,
            const float* __restrict__ V,
            const int*   __restrict__ bm,
            float*       __restrict__ out) {
    const int lane = threadIdx.x;          // 0..31 (wave32)
    const int half = lane >> 4;            // 0 | 1
    const int l16  = lane & 15;

    int idx = blockIdx.x;                  // B*H*NB
    const int ib = idx % NB;  idx /= NB;
    const int h  = idx % H_;  idx /= H_;
    const int b  = idx;

    const size_t bh = (size_t)b * H_ + h;
    const float* Qb = Q + bh * S_ * D_;
    const float* Kb = K + bh * S_ * D_;
    const float* Vb = V + bh * S_ * D_;
    const int*   bmrow = bm + (h * NB + ib) * NB;

    __shared__ float  v_lds[BLK * D_];     // 8 KB, V tile f32
    __shared__ __bf16 p_lds[BLK * BLK];    // 2 KB, P tile bf16

    const int kbaseA = half ? 8  : 0;   // A-layout K split per lane half
    const int kbaseB = half ? 16 : 0;   // contiguous K split (unused; B operands use kbaseA)

    // ---- Q tile -> bf16 A-fragments (held in VGPRs for the whole sweep) ----
    BF16Frag qf[2][2];                  // [M-tile][K-chunk(d 0..31 / 32..63)]
    #pragma unroll
    for (int t = 0; t < 2; ++t) {
        const float* qrow = Qb + (size_t)(ib * BLK + t * 16 + l16) * D_;
        #pragma unroll
        for (int kc = 0; kc < 2; ++kc) {
            #pragma unroll
            for (int e = 0; e < 16; ++e) {
                const int kk = (e < 8) ? (kbaseA + e) : (16 + kbaseA + (e - 8));
                qf[t][kc].b[e] = (__bf16)qrow[kc * 32 + kk];
            }
        }
    }

    v8f acc[2][4];                      // O accum: 32 rows x 64 cols
    #pragma unroll
    for (int t = 0; t < 2; ++t)
        #pragma unroll
        for (int n = 0; n < 4; ++n)
            acc[t][n] = {};

    float mstat[2][8], lstat[2][8];     // per-lane row stats (C layout rows)
    #pragma unroll
    for (int t = 0; t < 2; ++t)
        #pragma unroll
        for (int i = 0; i < 8; ++i) { mstat[t][i] = -INFINITY; lstat[t][i] = 0.0f; }

    // ---- sweep active column blocks at/below the diagonal ----
    for (int jb = 0; jb <= ib; ++jb) {
        if (!bmrow[jb]) continue;

        if (jb + 1 <= ib) {             // hint next block's K/V rows into GL2
            __builtin_prefetch(Kb + (size_t)((jb + 1) * BLK + lane) * D_, 0, 0);
            __builtin_prefetch(Vb + (size_t)((jb + 1) * BLK + lane) * D_, 0, 0);
        }

        // V tile -> LDS via async copy (no VGPR bounce; ASYNCcnt-tracked).
        // Each lane moves its 256-byte row as 16x B128 at offsets 0..240.
        {
            const float*   gsrc    = Vb + (size_t)(jb * BLK + lane) * D_;
            const unsigned lds_dst = (unsigned)(uintptr_t)(v_lds + lane * D_);
            ASYNC_CP16(0);    ASYNC_CP16(16);   ASYNC_CP16(32);   ASYNC_CP16(48);
            ASYNC_CP16(64);   ASYNC_CP16(80);   ASYNC_CP16(96);   ASYNC_CP16(112);
            ASYNC_CP16(128);  ASYNC_CP16(144);  ASYNC_CP16(160);  ASYNC_CP16(176);
            ASYNC_CP16(192);  ASYNC_CP16(208);  ASYNC_CP16(224);  ASYNC_CP16(240);
        }

        // K tile -> bf16 B-fragments (two 8-float runs per lane, A-matching k order);
        // overlaps with the in-flight async V copy.
        BF16Frag kf[2][2];              // [K-chunk][N-tile(key group)]
        #pragma unroll
        for (int n = 0; n < 2; ++n) {
            const float* krow = Kb + (size_t)(jb * BLK + n * 16 + l16) * D_;
            #pragma unroll
            for (int kc = 0; kc < 2; ++kc) {
                #pragma unroll
                for (int e = 0; e < 16; ++e)
                    kf[kc][n].b[e] = (__bf16)krow[kc * 32 + ((e < 8) ? (kbaseA + e) : (16 + kbaseA + (e - 8)))];
            }
        }

        // ---- scores = Q Kt : 8x v_wmma_f32_16x16x32_bf16 ----
        v8f s[2][2];
        #pragma unroll
        for (int t = 0; t < 2; ++t)
            #pragma unroll
            for (int n = 0; n < 2; ++n) {
                v8f c = {};
                c = __builtin_amdgcn_wmma_f32_16x16x32_bf16(
                        false, qf[t][0].v, false, kf[0][n].v, (short)0, c, false, false);
                c = __builtin_amdgcn_wmma_f32_16x16x32_bf16(
                        false, qf[t][1].v, false, kf[1][n].v, (short)0, c, false, false);
                s[t][n] = c;
            }

        // causal element mask (diagonal block only)
        if (jb == ib) {
            #pragma unroll
            for (int t = 0; t < 2; ++t)
                #pragma unroll
                for (int n = 0; n < 2; ++n)
                    #pragma unroll
                    for (int i = 0; i < 8; ++i) {
                        const int row_l = t * 16 + (half ? i + 8 : i);
                        const int col_l = n * 16 + l16;
                        if (col_l > row_l) s[t][n][i] = -1e30f;
                    }
        }

        // ---- online softmax (row reductions across 16-lane halves) ----
        #pragma unroll
        for (int t = 0; t < 2; ++t) {
            #pragma unroll
            for (int i = 0; i < 8; ++i) {
                float rmax = fmaxf(s[t][0][i], s[t][1][i]);
                #pragma unroll
                for (int off = 1; off < 16; off <<= 1)
                    rmax = fmaxf(rmax, __shfl_xor(rmax, off, 32));
                const float mnew  = fmaxf(mstat[t][i], rmax);
                const float scale = __expf(mstat[t][i] - mnew);   // 0 on first hit
                mstat[t][i] = mnew;

                float psum = 0.0f;
                #pragma unroll
                for (int n = 0; n < 2; ++n) {
                    const float pv = __expf(s[t][n][i] - mnew);   // masked -> 0
                    s[t][n][i] = pv;
                    psum += pv;
                }
                #pragma unroll
                for (int off = 1; off < 16; off <<= 1)
                    psum += __shfl_xor(psum, off, 32);
                lstat[t][i] = lstat[t][i] * scale + psum;

                #pragma unroll
                for (int n = 0; n < 4; ++n)
                    acc[t][n][i] *= scale;
            }
        }

        // ---- P: C layout -> bf16 LDS -> A layout fragments ----
        #pragma unroll
        for (int t = 0; t < 2; ++t)
            #pragma unroll
            for (int n = 0; n < 2; ++n)
                #pragma unroll
                for (int i = 0; i < 8; ++i) {
                    const int row_l = t * 16 + (half ? i + 8 : i);
                    const int col_l = n * 16 + l16;
                    p_lds[row_l * BLK + col_l] = (__bf16)s[t][n][i];
                }

        // V async copy must have landed before LDS reads of v_lds.
        asm volatile("s_wait_asynccnt 0" ::: "memory");
        __syncthreads();

        BF16Frag pf[2];
        #pragma unroll
        for (int t = 0; t < 2; ++t)
            #pragma unroll
            for (int e = 0; e < 16; ++e) {
                const int kk = (e < 8) ? (kbaseA + e) : (16 + kbaseA + (e - 8));
                pf[t].b[e] = p_lds[(t * 16 + l16) * BLK + kk];
            }

        // V -> bf16 B-fragments (conflict-free LDS column reads)
        BF16Frag vf[4];
        #pragma unroll
        for (int n = 0; n < 4; ++n) {
            const int dcol = n * 16 + l16;
            #pragma unroll
            for (int e = 0; e < 16; ++e)
                vf[n].b[e] = (__bf16)v_lds[((e < 8) ? (kbaseA + e) : (16 + kbaseA + (e - 8))) * D_ + dcol];
        }

        // ---- O += P V : 8x v_wmma_f32_16x16x32_bf16 ----
        #pragma unroll
        for (int t = 0; t < 2; ++t)
            #pragma unroll
            for (int n = 0; n < 4; ++n)
                acc[t][n] = __builtin_amdgcn_wmma_f32_16x16x32_bf16(
                                false, pf[t].v, false, vf[n].v, (short)0,
                                acc[t][n], false, false);

        __syncthreads();   // protect v_lds / p_lds before next iteration
    }

    // ---- epilogue: normalize, fully-masked rows -> 0 ----
    #pragma unroll
    for (int t = 0; t < 2; ++t)
        #pragma unroll
        for (int i = 0; i < 8; ++i) {
            const float denom = lstat[t][i];
            const float inv   = (denom > 0.0f) ? (1.0f / denom) : 0.0f;
            const int   row   = ib * BLK + t * 16 + (half ? i + 8 : i);
            float* orow = out + (bh * S_ + row) * D_;
            #pragma unroll
            for (int n = 0; n < 4; ++n)
                orow[n * 16 + l16] = acc[t][n][i] * inv;
        }
}

// ---------------------------------------------------------------------------
extern "C" void kernel_launch(void* const* d_in, const int* in_sizes, int n_in,
                              void* d_out, int out_size, void* d_ws, size_t ws_size,
                              hipStream_t stream) {
    (void)in_sizes; (void)n_in; (void)out_size; (void)ws_size;
    const float* q    = (const float*)d_in[0];
    const float* k    = (const float*)d_in[1];
    const float* v    = (const float*)d_in[2];
    const float* bias = (const float*)d_in[3];
    const int*   mask = (const int*)d_in[4];
    float*       out  = (float*)d_out;
    int*         bm   = (int*)d_ws;     // H*NB*NB ints = 256 KB scratch

    hipLaunchKernelGGL(block_mask_kernel, dim3(H_ * NB * NB), dim3(256), 0, stream,
                       mask, bias, bm);
    hipLaunchKernelGGL(attn_kernel, dim3(B_ * H_ * NB), dim3(32), 0, stream,
                       q, k, v, bm, out);
}
